// GameTheoryValues_75127567942222
// MI455X (gfx1250) — matrix operand order repaired and checked by reading, not compiled, not run on hardware
//
#include <hip/hip_runtime.h>
#include <stdint.h>

#define B_SZ 256
#define NSP  128
#define DM   768
#define KMC  25
#define KT   24   // 768/32 k-steps
#define NT   48   // 768/16 n-tiles

typedef __attribute__((ext_vector_type(16))) __bf16 v16bf;
typedef __attribute__((ext_vector_type(8)))  float  v8f;
typedef __attribute__((ext_vector_type(4)))  int    v4i;

#define GAS __attribute__((address_space(1)))
#define LAS __attribute__((address_space(3)))

// Native RNE conversion: lets the backend emit v_cvt_*bf16* instead of a
// 3-op manual round (bfe/add3/mov) per element.
__device__ __forceinline__ unsigned short f2bf(float f) {
    union { __bf16 h; unsigned short s; } u;
    u.h = (__bf16)f;
    return u.s;
}

union FragU { v16bf v; uint4 q[2]; unsigned short s[16]; };

// ---------------------------------------------------------------------------
// Kernel 0a: bf16 A-fragments of the coalition matrix (2 M-tiles x 4 K-steps).
// A 16-bit layout (ISA 7.12.2): lanes<16 -> M=lane, K = {0..7,16..23}+kt*32;
// lanes>=16 -> M=lane-16, K = {8..15,24..31}+kt*32.
__global__ void k_build_cfrag(const int* __restrict__ coal, unsigned short* __restrict__ cf) {
    int idx  = blockIdx.x * blockDim.x + threadIdx.x;   // < 8*512
    int i    = idx & 15;
    int lane = (idx >> 4) & 31;
    int frag = idx >> 9;                                 // mt*4 + kt
    int kt = frag & 3, mt = frag >> 2;
    int M = mt * 16 + (lane & 15);
    int K = kt * 32 + (lane >> 4) * 8 + (i < 8 ? i : 8 + i);
    unsigned short v = 0;
    if (M < KMC) v = (coal[M * NSP + K] & 1) ? 0x3F80 : 0;  // bf16(1.0)/bf16(0.0)
    cf[idx] = v;
}

// ---------------------------------------------------------------------------
// Kernel 0b: pre-swizzle w_v into B-fragment order (bf16).
// B 16-bit layout: lanes 0-15 hold K={0..7,16..23} (N=lane), lanes 16-31 hold K={8..15,24..31}.
// Stored half-major so GEMM does two fully coalesced b128 loads per fragment.
__global__ void k_swizzle_wv(const float* __restrict__ wv, unsigned short* __restrict__ out) {
    int idx  = blockIdx.x * blockDim.x + threadIdx.x;    // < 24*48*512
    int j    = idx & 7;
    int lane = (idx >> 3) & 31;
    int hf   = (idx >> 8) & 1;
    int frag = idx >> 9;                                  // kt*48 + nt
    int nt = frag % NT, kt = frag / NT;
    int K = kt * 32 + hf * 16 + (lane >> 4) * 8 + j;
    int N = nt * 16 + (lane & 15);
    out[idx] = f2bf(wv[K * DM + N]);
}

// ---------------------------------------------------------------------------
// Stage 1: s = c @ features[b] via WMMA (bf16 in, f32 acc), output bf16.
// One WG per batch, 4 waves: wave = (mtile<<1)|n-half. Also zeroes norm2.
__global__ void __launch_bounds__(128)
k_stage1(const float* __restrict__ feat, const uint4* __restrict__ cfragq,
         unsigned short* __restrict__ sbf, float* __restrict__ norm2) {
    const int b    = blockIdx.x;
    const int tid  = threadIdx.x;
    const int wave = tid >> 5, L = tid & 31;
    const int mt   = wave >> 1, half = wave & 1;
    const int r    = L & 15,  hi = L >> 4;

    if (tid < KMC) norm2[b * KMC + tid] = 0.0f;

    FragU A[4];
#pragma unroll
    for (int kt = 0; kt < 4; ++kt) {
        const uint4* fq = cfragq + ((size_t)((mt * 4 + kt) * 32 + L)) * 2;
        A[kt].q[0] = fq[0];
        A[kt].q[1] = fq[1];
    }

    const float* fb = feat + (size_t)b * NSP * DM;
    for (int nt = half * 24; nt < half * 24 + 24; ++nt) {
        const int n0 = nt * 16 + r;
        v8f acc = {0.f,0.f,0.f,0.f,0.f,0.f,0.f,0.f};
#pragma unroll
        for (int kt = 0; kt < 4; ++kt) {
            FragU Bu;
#pragma unroll
            for (int i = 0; i < 16; ++i) {
                int K = kt * 32 + hi * 8 + (i < 8 ? i : 8 + i);
                Bu.v[i] = (__bf16)fb[(size_t)K * DM + n0];   // native v_cvt
            }
            acc = __builtin_amdgcn_wmma_f32_16x16x32_bf16(
                      false, A[kt].v, false, Bu.v, (short)0, acc, false, false);
        }
        // C layout: VGPR i -> row M = i + hi*8, col N = n0
#pragma unroll
        for (int i = 0; i < 8; ++i) {
            int mrow = mt * 16 + hi * 8 + i;
            if (mrow < KMC)
                sbf[((size_t)(b * KMC + mrow)) * DM + n0] = f2bf(acc[i]);
        }
    }
}

// ---------------------------------------------------------------------------
// Stage 2: norm2 = rowwise ||s @ w_v||^2, fused (proj never materialized).
// 200 WGs x 4 waves; wave owns M-tile (wave>>1) and N-half (wave&1).
// A tiles staged in LDS with 16B-chunk XOR swizzle (chunk^row) -> conflict-free
// ds_load_b128 fragment reads. Tile fill uses GLOBAL_LOAD_ASYNC_TO_LDS_B128
// (ASYNCcnt) when the toolchain exposes it.
__global__ void __launch_bounds__(128)
k_gemm_norm(const unsigned short* __restrict__ sbf, const uint4* __restrict__ wvq,
            float* __restrict__ norm2) {
    __shared__ uint4 sAq[2 * 16 * 96];                    // 2 tiles * 16 rows * 96 chunks
    const int g = blockIdx.x;                             // 0..199 -> rows g*32..g*32+31
    const int tid = threadIdx.x;
    const int wave = tid >> 5, L = tid & 31;

    // cooperative global->LDS copy, 8 rows per wave, XOR-swizzled chunks
    const uint4* sq = (const uint4*)sbf;
    for (int rr = wave * 8; rr < wave * 8 + 8; ++rr) {
        const int row = g * 32 + rr;
        const int r = rr & 15;
        const int tb = (rr >> 4) * 1536 + r * 96;
        const uint4* src = sq + (size_t)row * 96;
#pragma unroll
        for (int it = 0; it < 3; ++it) {
            int c = it * 32 + L;
#if defined(__gfx1250__) && __has_builtin(__builtin_amdgcn_global_load_async_to_lds_b128)
            __builtin_amdgcn_global_load_async_to_lds_b128(
                (GAS v4i*)(src + c), (LAS v4i*)(&sAq[tb + (c ^ r)]), 0, 0);
#else
            sAq[tb + (c ^ r)] = src[c];
#endif
        }
    }
#if defined(__gfx1250__) && __has_builtin(__builtin_amdgcn_global_load_async_to_lds_b128)
#if __has_builtin(__builtin_amdgcn_s_wait_asynccnt)
    __builtin_amdgcn_s_wait_asynccnt(0);
#else
    asm volatile("s_wait_asynccnt 0x0" ::: "memory");
#endif
#endif
    __syncthreads();

    const int tile = wave >> 1, half = wave & 1;
    const int m0 = g * 32 + tile * 16;
    const int r = L & 15, hi = L >> 4;
    const int tb = tile * 1536 + r * 96;

    v8f nrm = {0.f,0.f,0.f,0.f,0.f,0.f,0.f,0.f};
    for (int nt = half * 24; nt < half * 24 + 24; ++nt) {
        v8f acc = {0.f,0.f,0.f,0.f,0.f,0.f,0.f,0.f};
#pragma unroll 4
        for (int kt = 0; kt < KT; ++kt) {
            const int c0 = kt * 4 + hi;
            FragU Au, Bu;
            Au.q[0] = sAq[tb + (c0 ^ r)];
            Au.q[1] = sAq[tb + ((c0 + 2) ^ r)];
            const uint4* fp = wvq + (size_t)(kt * NT + nt) * 64;
            Bu.q[0] = fp[L];                               // coalesced 512B
            Bu.q[1] = fp[32 + L];
            acc = __builtin_amdgcn_wmma_f32_16x16x32_bf16(
                      false, Au.v, false, Bu.v, (short)0, acc, false, false);
        }
#pragma unroll
        for (int i = 0; i < 8; ++i) nrm[i] += acc[i] * acc[i];
    }
    // reduce over the 16 N-lanes of each half; lane 0 -> rows 0..7, lane 16 -> rows 8..15
#pragma unroll
    for (int i = 0; i < 8; ++i) {
        float v = nrm[i];
        v += __shfl_xor(v, 1, 32);
        v += __shfl_xor(v, 2, 32);
        v += __shfl_xor(v, 4, 32);
        v += __shfl_xor(v, 8, 32);
        if (r == 0) atomicAdd(&norm2[m0 + hi * 8 + i], v);
    }
}

// ---------------------------------------------------------------------------
// Stage 3: cv = relu(sqrt(norm2)*scale); shapley = mean(cv|in) - mean(cv|out).
__global__ void k_shapley(const float* __restrict__ norm2, const int* __restrict__ coal,
                          const float* __restrict__ scale, float* __restrict__ out) {
    const int b = blockIdx.x;
    const int n = threadIdx.x;                             // 128
    __shared__ float scv[KMC];
    if (n < KMC) {
        float v = sqrtf(fmaxf(norm2[b * KMC + n], 0.0f)) * scale[0];
        scv[n] = fmaxf(v, 0.0f);
    }
    __syncthreads();
    float sw = 0.0f, total = 0.0f;
    int cnt = 0;
#pragma unroll
    for (int k = 0; k < KMC; ++k) {
        float cv = scv[k];
        int mi = coal[k * NSP + n] & 1;
        total += cv;
        if (mi) { sw += cv; ++cnt; }
    }
    float vw  = sw / (float)(cnt > 0 ? cnt : 1);
    float vwo = (total - sw) / (float)((KMC - cnt) > 0 ? (KMC - cnt) : 1);
    out[b * NSP + n] = (cnt > 0 && cnt < KMC) ? (vw - vwo) : 0.0f;
}

// ---------------------------------------------------------------------------
extern "C" void kernel_launch(void* const* d_in, const int* in_sizes, int n_in,
                              void* d_out, int out_size, void* d_ws, size_t ws_size,
                              hipStream_t stream) {
    const float* feat  = (const float*)d_in[0];
    const int*   coal  = (const int*)  d_in[1];
    const float* wv    = (const float*)d_in[2];
    const float* scale = (const float*)d_in[3];
    float* out = (float*)d_out;

    char* ws = (char*)d_ws;
    unsigned short* sbf   = (unsigned short*)ws;                         //  9,830,400 B
    unsigned short* wvs   = (unsigned short*)(ws + 9830400);             //  1,179,648 B
    float*          norm2 = (float*)(ws + 9830400 + 1179648);            //     25,600 B
    unsigned short* cfrag = (unsigned short*)(ws + 9830400 + 1179648 + 25600); // 8,192 B

    k_build_cfrag<<<16,   256, 0, stream>>>(coal, cfrag);
    k_swizzle_wv <<<2304, 256, 0, stream>>>(wv, wvs);
    k_stage1     <<<B_SZ, 128, 0, stream>>>(feat, (const uint4*)cfrag, sbf, norm2);
    k_gemm_norm  <<<200,  128, 0, stream>>>(sbf, (const uint4*)wvs, norm2);
    k_shapley    <<<B_SZ, 128, 0, stream>>>(norm2, coal, scale, out);
}
